// LSTMRegressor_68547678044465
// MI455X (gfx1250) — matrix operand order repaired and checked by reading, not compiled, not run on hardware
//
#include <hip/hip_runtime.h>

namespace {

constexpr int B_  = 2048;
constexpr int T_  = 512;
constexpr int D_  = 32;
constexpr int H_  = 64;
constexpr int G_  = 4 * H_;     // 256 gate columns
constexpr int MT  = 16;         // batch tile per workgroup
constexpr int NTHREADS = 256;   // 8 waves (wave32)
constexpr float LN_EPS = 1e-5f;

typedef __attribute__((ext_vector_type(16))) __bf16 v16bf;
typedef __attribute__((ext_vector_type(8)))  float  v8f;

struct U4x2 { uint4 lo, hi; };

__device__ __forceinline__ __bf16 f2bf(float f) {
  unsigned u = __builtin_bit_cast(unsigned, f);
  unsigned short s = (unsigned short)(u >> 16);
  return __builtin_bit_cast(__bf16, s);
}

__device__ __forceinline__ unsigned short f2bfbits(float f) {
  unsigned u = __builtin_bit_cast(unsigned, f);
  return (unsigned short)(u >> 16);
}

// pack two f32 -> two bf16 in one dword (truncation round)
__device__ __forceinline__ unsigned pack2bf(float lo, float hi) {
  unsigned ul = __builtin_bit_cast(unsigned, lo);
  unsigned uh = __builtin_bit_cast(unsigned, hi);
  return (uh & 0xffff0000u) | (ul >> 16);
}

__device__ __forceinline__ v16bf cvt16(float4 a, float4 b, float4 c, float4 d) {
  v16bf r;
  r[0]  = f2bf(a.x); r[1]  = f2bf(a.y); r[2]  = f2bf(a.z); r[3]  = f2bf(a.w);
  r[4]  = f2bf(b.x); r[5]  = f2bf(b.y); r[6]  = f2bf(b.z); r[7]  = f2bf(b.w);
  r[8]  = f2bf(c.x); r[9]  = f2bf(c.y); r[10] = f2bf(c.z); r[11] = f2bf(c.w);
  r[12] = f2bf(d.x); r[13] = f2bf(d.y); r[14] = f2bf(d.z); r[15] = f2bf(d.w);
  return r;
}

// B fragment for v_wmma_f32_16x16x32_bf16.
// gates = act * W^T, so B[k][n] = W[n][k]: per lane (n = ncol), K = k0..k0+7 and
// k0+16..k0+23 of W row n, k0 = 32-tile base + 8*(lane half).
__device__ __forceinline__ v16bf load_bfrag(const float* __restrict__ W, int ldk,
                                            int n, int kbase) {
  const int kb0 = (kbase & ~31) + ((kbase & 16) >> 1);
  const float4* p = reinterpret_cast<const float4*>(W + (size_t)n * ldk + kb0);
  return cvt16(p[0], p[1], p[4], p[5]);
}

// A fragment (16-bit, 16x32 K-tile kt) from LDS bf16 row-major [MT][ldk].
// Lanes 0-15: m=lane,  K = k0..k0+7 and k0+16..k0+23 with k0 = kt*32
// Lanes 16-31: m=lane-16, same pattern with k0 = kt*32 + 8
__device__ __forceinline__ v16bf load_afrag_lds(const unsigned short* hsh, int ldk,
                                                int lane, int kt) {
  int m  = lane & 15;
  int k0 = kt * 32 + ((lane >> 4) << 3);
  const unsigned short* row = hsh + m * ldk + k0;
  U4x2 u;
  u.lo = *reinterpret_cast<const uint4*>(row);       // K offsets 0..7
  u.hi = *reinterpret_cast<const uint4*>(row + 16);  // K offsets 16..23
  return __builtin_bit_cast(v16bf, u);
}

// Native v_rcp_f32 (1 instruction) instead of IEEE-correct divide (~9 instrs).
__device__ __forceinline__ float fast_rcp(float x) {
  return __builtin_amdgcn_rcpf(x);
}
__device__ __forceinline__ float sigmoidf_(float x) {
  return fast_rcp(1.0f + __expf(-x));
}
__device__ __forceinline__ float tanhf_(float x) {
  return 2.0f * sigmoidf_(2.0f * x) - 1.0f;
}

__device__ __forceinline__ v8f wmma_bf16(v16bf a, v16bf b, v8f c) {
  return __builtin_amdgcn_wmma_f32_16x16x32_bf16(
      /*neg_a=*/false, a, /*neg_b=*/false, b,
      /*c_mod=*/(short)0, c, /*reuse_a=*/false, /*reuse_b=*/false);
}

// ---------------- pre-pass: x f32 -> bf16 (pure bandwidth) ----------------
__global__ __launch_bounds__(256, 1) void cvt_x_bf16(const float* __restrict__ x,
                                                     unsigned* __restrict__ xbf /*uint-packed*/) {
  // 8 floats -> one uint4 of 8 bf16 per thread-iteration
  const size_t n8 = (size_t)B_ * T_ * D_ / 8;
  size_t i = (size_t)blockIdx.x * blockDim.x + threadIdx.x;
  const float4* x4 = reinterpret_cast<const float4*>(x);
  uint4* o4 = reinterpret_cast<uint4*>(xbf);
  for (; i < n8; i += (size_t)gridDim.x * blockDim.x) {
    float4 f0 = x4[2 * i];
    float4 f1 = x4[2 * i + 1];
    uint4 u;
    u.x = pack2bf(f0.x, f0.y);
    u.y = pack2bf(f0.z, f0.w);
    u.z = pack2bf(f1.x, f1.y);
    u.w = pack2bf(f1.z, f1.w);
    o4[i] = u;
  }
}

// ---------------- fused 2-layer LSTM scan + LN + FC ----------------
template <bool XPRE /* x already bf16 in workspace */>
__global__ __launch_bounds__(NTHREADS, 1) void lstm2_fused(
    const float* __restrict__ x,
    const unsigned short* __restrict__ xbf,
    const float* __restrict__ Wih0, const float* __restrict__ Whh0,
    const float* __restrict__ bih0, const float* __restrict__ bhh0,
    const float* __restrict__ Wih1, const float* __restrict__ Whh1,
    const float* __restrict__ bih1, const float* __restrict__ bhh1,
    const float* __restrict__ gamma, const float* __restrict__ beta,
    const float* __restrict__ fcw,  const float* __restrict__ fcb,
    float* __restrict__ out) {
  __shared__ float          gates[MT * G_];          // 16 KB, reused by both layers
  __shared__ unsigned short h0sh[MT * H_];           // 2 KB  (bf16 h state, layer 0)
  __shared__ unsigned short h1sh[MT * H_];           // 2 KB  (bf16 h state, layer 1)
  __shared__ float          hlast[MT * H_];          // 4 KB  (f32 h1 at t = T-1)

  const int tid   = threadIdx.x;
  const int lane  = tid & 31;
  const int wave  = tid >> 5;
  const int bBase = blockIdx.x * MT;

  // ---- preload all weight B-fragments into registers (resident for all 512 steps) ----
  const int nt0 = wave * 32;                 // this wave owns N-tiles nt0 and nt0+16
  const int nloc = lane & 15;
  const int kb   = (lane >> 4) * 16;         // lanes 0-15 -> K 0..7,16..23; lanes 16-31 -> K 8..15,24..31

  v16bf Bih0[2], Bhh0[2][2], Bih1[2][2], Bhh1[2][2];
  float bias0[2], bias1[2];
#pragma unroll
  for (int s = 0; s < 2; ++s) {
    const int n = nt0 + s * 16 + nloc;
    Bih0[s]    = load_bfrag(Wih0, D_, n, kb);          // K-tile 0 (K=32 total)
    Bhh0[s][0] = load_bfrag(Whh0, H_, n, kb);
    Bhh0[s][1] = load_bfrag(Whh0, H_, n, kb + 32);
    Bih1[s][0] = load_bfrag(Wih1, H_, n, kb);
    Bih1[s][1] = load_bfrag(Wih1, H_, n, kb + 32);
    Bhh1[s][0] = load_bfrag(Whh1, H_, n, kb);
    Bhh1[s][1] = load_bfrag(Whh1, H_, n, kb + 32);
    bias0[s] = bih0[n] + bhh0[n];
    bias1[s] = bih1[n] + bhh1[n];
  }

  // ---- init recurrent state ----
  for (int i = tid; i < MT * H_; i += NTHREADS) { h0sh[i] = 0; h1sh[i] = 0; }
  float c0r[4], c1r[4];
#pragma unroll
  for (int k = 0; k < 4; ++k) { c0r[k] = 0.0f; c1r[k] = 0.0f; }
  __syncthreads();

  // C-matrix layout: VGPR r -> (M = r + 8*(lane>=16), N = ntile + (lane&15))
  const int crow  = (lane >> 4) * 8;
  const int ccol0 = nt0 + nloc;

  // per-lane x row pointers (lane's A-fragment row m = lane&15)
  const size_t xoff = ((size_t)(bBase + nloc) * T_) * D_ + ((lane >> 4) << 3);
  const float*          xrow0f = x   + xoff;
  const unsigned short* xrow0b = xbf + xoff;

  for (int t = 0; t < T_; ++t) {
    // ===================== layer 0 gates =====================
    v16bf xa;
    if constexpr (XPRE) {
      const unsigned short* xr = xrow0b + (size_t)t * D_;
      __builtin_prefetch(xr + D_, 0, 0);   // global_prefetch_b8: next timestep slice
      U4x2 u;
      u.lo = *reinterpret_cast<const uint4*>(xr);        // K 0..7   (8 bf16)
      u.hi = *reinterpret_cast<const uint4*>(xr + 16);   // K 16..23 (8 bf16)
      xa = __builtin_bit_cast(v16bf, u);
    } else {
      const float* xr = xrow0f + (size_t)t * D_;
      __builtin_prefetch(xr + D_, 0, 0);
      float4 xa0 = *reinterpret_cast<const float4*>(xr);
      float4 xa1 = *reinterpret_cast<const float4*>(xr + 4);
      float4 xa2 = *reinterpret_cast<const float4*>(xr + 16);
      float4 xa3 = *reinterpret_cast<const float4*>(xr + 20);
      xa = cvt16(xa0, xa1, xa2, xa3);
    }
    v16bf ha0 = load_afrag_lds(h0sh, H_, lane, 0);
    v16bf ha1 = load_afrag_lds(h0sh, H_, lane, 1);

#pragma unroll
    for (int s = 0; s < 2; ++s) {
      v8f acc;
#pragma unroll
      for (int r = 0; r < 8; ++r) acc[r] = bias0[s];
      acc = wmma_bf16(xa,  Bih0[s],    acc);
      acc = wmma_bf16(ha0, Bhh0[s][0], acc);
      acc = wmma_bf16(ha1, Bhh0[s][1], acc);
      const int col = ccol0 + s * 16;
#pragma unroll
      for (int r = 0; r < 8; ++r) gates[(crow + r) * G_ + col] = acc[r];
    }
    __syncthreads();

    // ===================== layer 0 elementwise =====================
#pragma unroll
    for (int k = 0; k < 4; ++k) {
      const int e  = tid + k * NTHREADS;
      const int m  = e >> 6;
      const int nh = e & 63;
      const float* gm = gates + m * G_;
      float ig = sigmoidf_(gm[nh]);
      float fg = sigmoidf_(gm[64 + nh]);
      float gg = tanhf_   (gm[128 + nh]);
      float og = sigmoidf_(gm[192 + nh]);
      float c  = fg * c0r[k] + ig * gg;
      c0r[k]   = c;
      float h  = og * tanhf_(c);
      h0sh[m * H_ + nh] = f2bfbits(h);
    }
    __syncthreads();

    // ===================== layer 1 gates =====================
    v16bf ia0 = load_afrag_lds(h0sh, H_, lane, 0);
    v16bf ia1 = load_afrag_lds(h0sh, H_, lane, 1);
    v16bf ja0 = load_afrag_lds(h1sh, H_, lane, 0);
    v16bf ja1 = load_afrag_lds(h1sh, H_, lane, 1);

#pragma unroll
    for (int s = 0; s < 2; ++s) {
      v8f acc;
#pragma unroll
      for (int r = 0; r < 8; ++r) acc[r] = bias1[s];
      acc = wmma_bf16(ia0, Bih1[s][0], acc);
      acc = wmma_bf16(ia1, Bih1[s][1], acc);
      acc = wmma_bf16(ja0, Bhh1[s][0], acc);
      acc = wmma_bf16(ja1, Bhh1[s][1], acc);
      const int col = ccol0 + s * 16;
#pragma unroll
      for (int r = 0; r < 8; ++r) gates[(crow + r) * G_ + col] = acc[r];
    }
    __syncthreads();

    // ===================== layer 1 elementwise =====================
#pragma unroll
    for (int k = 0; k < 4; ++k) {
      const int e  = tid + k * NTHREADS;
      const int m  = e >> 6;
      const int nh = e & 63;
      const float* gm = gates + m * G_;
      float ig = sigmoidf_(gm[nh]);
      float fg = sigmoidf_(gm[64 + nh]);
      float gg = tanhf_   (gm[128 + nh]);
      float og = sigmoidf_(gm[192 + nh]);
      float c  = fg * c1r[k] + ig * gg;
      c1r[k]   = c;
      float h  = og * tanhf_(c);
      h1sh[m * H_ + nh] = f2bfbits(h);
      if (t == T_ - 1) hlast[m * H_ + nh] = h;   // keep f32 precision for LN+FC
    }
    __syncthreads();
  }

  // ===================== LayerNorm + FC on last h1 =====================
  if (tid < MT) {
    const float* hr = hlast + tid * H_;
    float mu = 0.0f;
#pragma unroll 8
    for (int j = 0; j < H_; ++j) mu += hr[j];
    mu *= (1.0f / H_);
    float var = 0.0f;
#pragma unroll 8
    for (int j = 0; j < H_; ++j) { float d = hr[j] - mu; var += d * d; }
    var *= (1.0f / H_);
    const float inv = rsqrtf(var + LN_EPS);
    float acc = fcb[0];
#pragma unroll 8
    for (int j = 0; j < H_; ++j)
      acc += ((hr[j] - mu) * inv * gamma[j] + beta[j]) * fcw[j];
    out[bBase + tid] = acc;
  }
}

}  // namespace

extern "C" void kernel_launch(void* const* d_in, const int* in_sizes, int n_in,
                              void* d_out, int out_size, void* d_ws, size_t ws_size,
                              hipStream_t stream) {
  (void)in_sizes; (void)n_in; (void)out_size;
  const float* x     = reinterpret_cast<const float*>(d_in[0]);
  const float* Wih0  = reinterpret_cast<const float*>(d_in[1]);
  const float* Whh0  = reinterpret_cast<const float*>(d_in[2]);
  const float* bih0  = reinterpret_cast<const float*>(d_in[3]);
  const float* bhh0  = reinterpret_cast<const float*>(d_in[4]);
  const float* Wih1  = reinterpret_cast<const float*>(d_in[5]);
  const float* Whh1  = reinterpret_cast<const float*>(d_in[6]);
  const float* bih1  = reinterpret_cast<const float*>(d_in[7]);
  const float* bhh1  = reinterpret_cast<const float*>(d_in[8]);
  const float* gamma = reinterpret_cast<const float*>(d_in[9]);
  const float* beta  = reinterpret_cast<const float*>(d_in[10]);
  const float* fcw   = reinterpret_cast<const float*>(d_in[11]);
  const float* fcb   = reinterpret_cast<const float*>(d_in[12]);
  float* out = reinterpret_cast<float*>(d_out);

  dim3 grid(B_ / MT);     // 128 workgroups, one 16-row batch tile each
  dim3 block(NTHREADS);   // 8 wave32 waves

  const size_t xbf_bytes = (size_t)B_ * T_ * D_ * sizeof(unsigned short);  // 64 MB
  if (ws_size >= xbf_bytes) {
    // one-time bandwidth pre-pass: x -> bf16 (removes per-step cvt VALU from
    // the recurrent kernel and halves its x traffic)
    unsigned* xbf = reinterpret_cast<unsigned*>(d_ws);
    const size_t n8 = (size_t)B_ * T_ * D_ / 8;
    int cgrid = (int)((n8 + 255) / 256);
    if (cgrid > 8192) cgrid = 8192;  // grid-stride
    cvt_x_bf16<<<cgrid, 256, 0, stream>>>(x, xbf);
    lstm2_fused<true><<<grid, block, 0, stream>>>(
        x, reinterpret_cast<const unsigned short*>(d_ws),
        Wih0, Whh0, bih0, bhh0, Wih1, Whh1, bih1, bhh1,
        gamma, beta, fcw, fcb, out);
  } else {
    lstm2_fused<false><<<grid, block, 0, stream>>>(
        x, nullptr,
        Wih0, Whh0, bih0, bhh0, Wih1, Whh1, bih1, bhh1,
        gamma, beta, fcw, fcb, out);
  }
}
